// DecoderOnlyTransformer_33097017983498
// MI455X (gfx1250) — matrix operand order repaired and checked by reading, not compiled, not run on hardware
//
#include <hip/hip_runtime.h>
#include <math.h>

typedef __bf16 bf16;
typedef __bf16 v16bf __attribute__((ext_vector_type(16)));
typedef __bf16 bf2   __attribute__((ext_vector_type(2)));
typedef __bf16 bf8v  __attribute__((ext_vector_type(8)));
typedef float  v8f   __attribute__((ext_vector_type(8)));

#define WMMA_BF16(a, b, c) \
  __builtin_amdgcn_wmma_f32_16x16x32_bf16(false, (a), false, (b), (short)0, (c), false, false)

// Build a v16bf fragment from two contiguous 16-byte runs.
// A-frag (16x32): lo = base + half*8,  hi = base + 16 + half*8   (ISA 7.12.2)
// B-frag (32x16): lo = base + half*8,  hi = base + 16 + half*8
__device__ __forceinline__ v16bf frag_from(const bf16* lo, const bf16* hi) {
  union { v16bf v; bf8v h[2]; } u;
  u.h[0] = *(const bf8v*)lo;
  u.h[1] = *(const bf8v*)hi;
  return u.v;
}

// CDNA5 async global->LDS copy (16 bytes per lane), tracked by ASYNCcnt.
__device__ __forceinline__ void async_copy_b128(const bf16* gsrc, void* lds_dst) {
  unsigned loff = (unsigned)(size_t)lds_dst;   // low 32 bits = LDS offset
  asm volatile("global_load_async_to_lds_b128 %0, %1, off"
               :: "v"(loff), "v"((const void*)gsrc) : "memory");
}
__device__ __forceinline__ void wait_async0() {
  asm volatile("s_wait_asynccnt 0" ::: "memory");
}

// 16-lane XOR butterflies on the VALU via DPP16 ROW_XMASK (0x160|mask):
// keeps softmax reductions out of the LDS pipeline (no ds_bpermute).
template <int MASK>
__device__ __forceinline__ float xor_max16(float x) {
  int xi = __builtin_bit_cast(int, x);
  int yi = __builtin_amdgcn_update_dpp(xi, xi, 0x160 + MASK, 0xf, 0xf, true);
  return fmaxf(x, __builtin_bit_cast(float, yi));
}
template <int MASK>
__device__ __forceinline__ float xor_add16(float x) {
  int xi = __builtin_bit_cast(int, x);
  int yi = __builtin_amdgcn_update_dpp(xi, xi, 0x160 + MASK, 0xf, 0xf, true);
  return x + __builtin_bit_cast(float, yi);
}
__device__ __forceinline__ float red_max16(float x) {
  x = xor_max16<1>(x); x = xor_max16<2>(x);
  x = xor_max16<4>(x); x = xor_max16<8>(x);
  return x;
}
__device__ __forceinline__ float red_add16(float x) {
  x = xor_add16<1>(x); x = xor_add16<2>(x);
  x = xor_add16<4>(x); x = xor_add16<8>(x);
  return x;
}

// ---------------------------------------------------------------------------
// Embedding + positional encoding: x[b,s,:] = emb[tok]*sqrt(D) + pe[s,:]
// ---------------------------------------------------------------------------
__global__ __launch_bounds__(256)
void embed_kernel(const int* __restrict__ tokens, const float* __restrict__ emb,
                  const float* __restrict__ pe, float* __restrict__ x) {
  const int D = 1024, S = 1024;
  int row = blockIdx.x;          // b*S + s
  int s = row & (S - 1);
  int tok = tokens[row];
  const float scale = 32.0f;     // sqrt(1024)
  for (int i = threadIdx.x; i < D; i += 256)
    x[(size_t)row * D + i] = emb[(size_t)tok * D + i] * scale + pe[(size_t)s * D + i];
}

// ---------------------------------------------------------------------------
// LayerNorm over D=1024, fp32 in -> bf16 out. One block per row.
// ---------------------------------------------------------------------------
__global__ __launch_bounds__(256)
void layernorm_kernel(const float* __restrict__ x, const float* __restrict__ g,
                      const float* __restrict__ b, bf16* __restrict__ out) {
  const int D = 1024;
  int row = blockIdx.x;
  int t = threadIdx.x;
  const float* xr = x + (size_t)row * D;
  float vals[4];
  float s = 0.0f, s2 = 0.0f;
#pragma unroll
  for (int i = 0; i < 4; ++i) {
    float f = xr[t + i * 256];
    vals[i] = f; s += f; s2 += f * f;
  }
  __shared__ float r1[256], r2[256];
  r1[t] = s; r2[t] = s2;
  __syncthreads();
  for (int off = 128; off > 0; off >>= 1) {
    if (t < off) { r1[t] += r1[t + off]; r2[t] += r2[t + off]; }
    __syncthreads();
  }
  float mean = r1[0] * (1.0f / D);
  float var  = r2[0] * (1.0f / D) - mean * mean;
  float inv  = rsqrtf(var + 1e-5f);
#pragma unroll
  for (int i = 0; i < 4; ++i) {
    int c = t + i * 256;
    out[(size_t)row * D + c] = (bf16)((vals[i] - mean) * inv * g[c] + b[c]);
  }
}

// ---------------------------------------------------------------------------
// GEMM: C(MxN) = A(MxK, bf16) @ W(KxN, fp32 -> bf16 in LDS) [+bias]
// Block tile 64x256, K-step 32, DOUBLE-BUFFERED LDS pipeline:
//   steady state = { stage(next buf): async A b128 + W f32->bf16 pair stores;
//                    compute(cur buf): 2 A-frags x 4 B-frags = 8 WMMAs;
//                    s_wait_asynccnt 0; one barrier }.
// ---------------------------------------------------------------------------
template <bool OUT_F32, bool RESID, bool GELU>
__global__ __launch_bounds__(256)
void gemm_wmma(const bf16* __restrict__ A, const float* __restrict__ W,
               const float* __restrict__ bias, const float* __restrict__ resid,
               void* __restrict__ outp, int M, int N, int K) {
  __shared__ bf16 sA[2][64][40];    // [buf][m][k], 16B-aligned runs
  __shared__ bf16 sW[2][256][40];   // [buf][n][k] transposed
  int tid = threadIdx.x;
  int n0 = blockIdx.x * 256;
  int m0 = blockIdx.y * 64;
  int lane = tid & 31, wid = tid >> 5;
  int lm = lane & 15, lh = lane >> 4;
  int msub = (wid >> 2) * 16;    // 0 or 16; second frag at +32
  int nsub = (wid & 3) * 64;

  v8f acc[2][4] = {};

  // Per-thread staging coordinates: A tile 64x32 bf16 = 8 bf16 (16B) / thread
  int ea = tid * 8;
  int am = ea >> 5, ak = ea & 31;

  auto stageA = [&](int k0, int bufi) {
    async_copy_b128(A + (size_t)(m0 + am) * K + k0 + ak, &sA[bufi][am][ak]);
  };
  auto stageW = [&](int k0, int bufi) {
#pragma unroll
    for (int i = 0; i < 4; ++i) {
      int u = i * 256 + tid;        // 1024 units: 2 k-rows x 4 n each
      int kp = u >> 6;              // 0..15  (kk = 2*kp)
      int n = (u & 63) * 4;
      const float* wp = W + (size_t)(k0 + 2 * kp) * N + n0 + n;
      float4 w0 = *(const float4*)wp;
      float4 w1 = *(const float4*)(wp + N);
      bf2 t0 = {(bf16)w0.x, (bf16)w1.x};
      bf2 t1 = {(bf16)w0.y, (bf16)w1.y};
      bf2 t2 = {(bf16)w0.z, (bf16)w1.z};
      bf2 t3 = {(bf16)w0.w, (bf16)w1.w};
      *(bf2*)&sW[bufi][n + 0][2 * kp] = t0;
      *(bf2*)&sW[bufi][n + 1][2 * kp] = t1;
      *(bf2*)&sW[bufi][n + 2][2 * kp] = t2;
      *(bf2*)&sW[bufi][n + 3][2 * kp] = t3;
    }
  };

  // Prologue: fill buffer 0
  stageA(0, 0);
  stageW(0, 0);
  wait_async0();
  __syncthreads();

  int nIt = K >> 5;
  for (int it = 0; it < nIt; ++it) {
    int cur = it & 1;
    int k0 = it * 32;

    // Issue next tile's staging before computing on current tile
    if (it + 1 < nIt) {
      stageA(k0 + 32, cur ^ 1);
      stageW(k0 + 32, cur ^ 1);
      if (it + 2 < nIt)
        __builtin_prefetch(W + (size_t)(k0 + 64 + (tid >> 6)) * N + n0 + (tid & 63) * 4, 0, 1);
    }

    const bf16* arow0 = &sA[cur][msub + lm][0];
    const bf16* arow1 = &sA[cur][msub + 32 + lm][0];
    v16bf af0 = frag_from(arow0 + lh * 8, arow0 + 16 + lh * 8);
    v16bf af1 = frag_from(arow1 + lh * 8, arow1 + 16 + lh * 8);
#pragma unroll
    for (int t = 0; t < 4; ++t) {
      const bf16* brow = &sW[cur][nsub + t * 16 + lm][0];
      v16bf bw = frag_from(brow + lh * 8, brow + 16 + lh * 8);
      acc[0][t] = WMMA_BF16(af0, bw, acc[0][t]);
      acc[1][t] = WMMA_BF16(af1, bw, acc[1][t]);
    }

    wait_async0();     // next tile's async-A landed (in our wave)
    __syncthreads();   // all waves done writing next buf / reading cur buf
  }

#pragma unroll
  for (int sub = 0; sub < 2; ++sub) {
#pragma unroll
    for (int t = 0; t < 4; ++t) {
      int col = n0 + nsub + t * 16 + lm;
      float bv = bias ? bias[col] : 0.0f;
#pragma unroll
      for (int r = 0; r < 8; ++r) {
        int row = m0 + msub + sub * 32 + r + 8 * lh;
        size_t idx = (size_t)row * N + col;
        float vv = acc[sub][t][r] + bv;
        if constexpr (RESID) vv += resid[idx];
        if constexpr (GELU)  vv = 0.5f * vv * (1.0f + erff(vv * 0.70710678118f));
        if constexpr (OUT_F32) ((float*)outp)[idx] = vv;
        else                   ((bf16*)outp)[idx] = (bf16)vv;
      }
    }
  }
}

// ---------------------------------------------------------------------------
// Flash attention: one wave per (b, h, 16-query tile). Streams 32-key tiles.
// Q·Kt and P·V through WMMA; online softmax reductions via DPP row_xmask;
// P tile round-trips per-wave LDS to convert C-layout -> A-fragment layout.
// ---------------------------------------------------------------------------
__global__ __launch_bounds__(256)
void flash_attn_kernel(const bf16* __restrict__ q, const bf16* __restrict__ k,
                       const bf16* __restrict__ v, const int* __restrict__ tokens,
                       bf16* __restrict__ ctx) {
  const int S = 1024, H = 16, DK = 64;
  __shared__ bf16 sP[8][16][40];   // per-wave P tile (16 q-rows x 32 keys)
  __shared__ bf16 sVt[8][64][40];  // per-wave V tile transposed [dk][key]
  int tid = threadIdx.x, wid = tid >> 5, lane = tid & 31;
  int lm = lane & 15, lh = lane >> 4;
  int gw = blockIdx.x * 8 + wid;
  int qt = gw & 63;          // S/16 = 64 query tiles
  int h  = (gw >> 6) & 15;
  int b  = gw >> 10;

  // Q fragments: a0 covers dk 0..31, a1 covers dk 32..63 (A-frag runs)
  int qrow_i = qt * 16 + lm;
  const bf16* qrow = q + (((size_t)b * S + qrow_i) * H + h) * DK;
  v16bf a0 = frag_from(qrow + lh * 8,      qrow + 16 + lh * 8);
  v16bf a1 = frag_from(qrow + 32 + lh * 8, qrow + 48 + lh * 8);

  v8f o0 = {}, o1 = {}, o2 = {}, o3 = {};
  float mrow[8], lrow[8];
#pragma unroll
  for (int r = 0; r < 8; ++r) { mrow[r] = -1e30f; lrow[r] = 0.0f; }

  int qmax = qt * 16 + 15;
  int nk = (qmax + 32) >> 5;   // ceil((qmax+1)/32) causal key tiles

  for (int it = 0; it < nk; ++it) {
    int kb = it * 32;

    // Stage V tile transposed: lane owns a key PAIR and a 32-wide dk half;
    // packs (key0,key1) bf16 pairs -> ds_store_b32
    {
      int kp = lane & 15, hd = lane >> 4;
      int key0 = kb + 2 * kp;
      const bf16* v0p = v + (((size_t)b * S + key0) * H + h) * DK + hd * 32;
      const bf16* v1p = v0p + H * DK;   // next key row
#pragma unroll
      for (int j = 0; j < 32; j += 8) {
        bf8v x0 = *(const bf8v*)(v0p + j);
        bf8v x1 = *(const bf8v*)(v1p + j);
#pragma unroll
        for (int e = 0; e < 8; ++e) {
          bf2 t = {x0[e], x1[e]};
          *(bf2*)&sVt[wid][hd * 32 + j + e][2 * kp] = t;
        }
      }
    }

    // K B-fragments straight from global (two b128 runs each)
    int key0 = kb + lm, key1 = kb + 16 + lm;
    const bf16* kr0 = k + (((size_t)b * S + key0) * H + h) * DK;
    const bf16* kr1 = k + (((size_t)b * S + key1) * H + h) * DK;
    v16bf b00 = frag_from(kr0 + lh * 8,      kr0 + 16 + lh * 8);
    v16bf b01 = frag_from(kr0 + 32 + lh * 8, kr0 + 48 + lh * 8);
    v16bf b10 = frag_from(kr1 + lh * 8,      kr1 + 16 + lh * 8);
    v16bf b11 = frag_from(kr1 + 32 + lh * 8, kr1 + 48 + lh * 8);

    v8f c0 = {}, c1 = {};
    c0 = WMMA_BF16(a0, b00, c0);
    c0 = WMMA_BF16(a1, b01, c0);
    c1 = WMMA_BF16(a0, b10, c1);
    c1 = WMMA_BF16(a1, b11, c1);

    bool tok0 = tokens[b * S + key0] != 0;
    bool tok1 = tokens[b * S + key1] != 0;

#pragma unroll
    for (int r = 0; r < 8; ++r) {
      int qr = qt * 16 + r + 8 * lh;
      float s0 = c0[r] * 0.125f;   // 1/sqrt(64)
      float s1 = c1[r] * 0.125f;
      if (!(tok0 && key0 <= qr)) s0 = -1e9f;
      if (!(tok1 && key1 <= qr)) s1 = -1e9f;

      float mt = red_max16(fmaxf(s0, s1));           // DPP butterfly, no LDS
      float mnew = fmaxf(mrow[r], mt);
      float alpha = __expf(mrow[r] - mnew);
      mrow[r] = mnew;
      float p0 = __expf(s0 - mnew);
      float p1 = __expf(s1 - mnew);
      float rs = red_add16(p0 + p1);                 // DPP butterfly, no LDS
      lrow[r] = lrow[r] * alpha + rs;
      o0[r] *= alpha; o1[r] *= alpha; o2[r] *= alpha; o3[r] *= alpha;
      sP[wid][r + 8 * lh][lm]      = (bf16)p0;
      sP[wid][r + 8 * lh][lm + 16] = (bf16)p1;
    }

    // Same-wave DS RAW: explicit wait before fragment reload from LDS
    asm volatile("s_wait_dscnt 0" ::: "memory");

    const bf16* prow = &sP[wid][lm][0];
    v16bf pa = frag_from(prow + lh * 8, prow + 16 + lh * 8);
    const bf16* vr0 = &sVt[wid][ 0 + lm][0];
    const bf16* vr1 = &sVt[wid][16 + lm][0];
    const bf16* vr2 = &sVt[wid][32 + lm][0];
    const bf16* vr3 = &sVt[wid][48 + lm][0];
    v16bf bv0 = frag_from(vr0 + lh * 8, vr0 + 16 + lh * 8);
    v16bf bv1 = frag_from(vr1 + lh * 8, vr1 + 16 + lh * 8);
    v16bf bv2 = frag_from(vr2 + lh * 8, vr2 + 16 + lh * 8);
    v16bf bv3 = frag_from(vr3 + lh * 8, vr3 + 16 + lh * 8);
    o0 = WMMA_BF16(pa, bv0, o0);
    o1 = WMMA_BF16(pa, bv1, o1);
    o2 = WMMA_BF16(pa, bv2, o2);
    o3 = WMMA_BF16(pa, bv3, o3);
  }

#pragma unroll
  for (int r = 0; r < 8; ++r) {
    float inv = 1.0f / lrow[r];
    int qr = qt * 16 + r + 8 * lh;
    bf16* crow = ctx + (((size_t)b * S + qr) * H + h) * DK;
    crow[ 0 + lm] = (bf16)(o0[r] * inv);
    crow[16 + lm] = (bf16)(o1[r] * inv);
    crow[32 + lm] = (bf16)(o2[r] * inv);
    crow[48 + lm] = (bf16)(o3[r] * inv);
  }
}

// ---------------------------------------------------------------------------
extern "C" void kernel_launch(void* const* d_in, const int* in_sizes, int n_in,
                              void* d_out, int out_size, void* d_ws, size_t ws_size,
                              hipStream_t stream) {
  const int*   tokens = (const int*)d_in[0];
  const float* emb    = (const float*)d_in[1];
  const float* pe     = (const float*)d_in[2];
  const float* Wq     = (const float*)d_in[3];
  const float* Wk     = (const float*)d_in[4];
  const float* Wv     = (const float*)d_in[5];
  const float* Wo     = (const float*)d_in[6];
  const float* bo     = (const float*)d_in[7];
  const float* g1     = (const float*)d_in[8];
  const float* be1    = (const float*)d_in[9];
  const float* g2     = (const float*)d_in[10];
  const float* be2    = (const float*)d_in[11];
  const float* W1     = (const float*)d_in[12];
  const float* b1     = (const float*)d_in[13];
  const float* W2     = (const float*)d_in[14];
  const float* b2     = (const float*)d_in[15];
  const float* gf     = (const float*)d_in[16];
  const float* bfp    = (const float*)d_in[17];
  const float* Wout   = (const float*)d_in[18];
  const float* bout   = (const float*)d_in[19];

  const int Dm = 1024, Fm = 4096, Vm = 32000, Mm = 2048, Lm = 6;

  char* ws = (char*)d_ws;
  float* x    = (float*)ws;  ws += (size_t)Mm * Dm * 4;
  bf16* hbuf  = (bf16*)ws;   ws += (size_t)Mm * Dm * 2;
  bf16* qbuf  = (bf16*)ws;   ws += (size_t)Mm * Dm * 2;
  bf16* kbuf  = (bf16*)ws;   ws += (size_t)Mm * Dm * 2;
  bf16* vbuf  = (bf16*)ws;   ws += (size_t)Mm * Dm * 2;
  bf16* cbuf  = (bf16*)ws;   ws += (size_t)Mm * Dm * 2;
  bf16* ffn   = (bf16*)ws;   ws += (size_t)Mm * Fm * 2;

  dim3 blk(256);
  dim3 gD(Dm / 256, Mm / 64);   //  4 x 32
  dim3 gF(Fm / 256, Mm / 64);   // 16 x 32
  dim3 gV(Vm / 256, Mm / 64);   // 125 x 32

  embed_kernel<<<Mm, blk, 0, stream>>>(tokens, emb, pe, x);

  for (int l = 0; l < Lm; ++l) {
    layernorm_kernel<<<Mm, blk, 0, stream>>>(x, g1 + (size_t)l * Dm, be1 + (size_t)l * Dm, hbuf);
    gemm_wmma<false, false, false><<<gD, blk, 0, stream>>>(
        hbuf, Wq + (size_t)l * Dm * Dm, nullptr, nullptr, qbuf, Mm, Dm, Dm);
    gemm_wmma<false, false, false><<<gD, blk, 0, stream>>>(
        hbuf, Wk + (size_t)l * Dm * Dm, nullptr, nullptr, kbuf, Mm, Dm, Dm);
    gemm_wmma<false, false, false><<<gD, blk, 0, stream>>>(
        hbuf, Wv + (size_t)l * Dm * Dm, nullptr, nullptr, vbuf, Mm, Dm, Dm);
    flash_attn_kernel<<<256, blk, 0, stream>>>(qbuf, kbuf, vbuf, tokens, cbuf);
    gemm_wmma<true, true, false><<<gD, blk, 0, stream>>>(
        cbuf, Wo + (size_t)l * Dm * Dm, bo + (size_t)l * Dm, x, x, Mm, Dm, Dm);
    layernorm_kernel<<<Mm, blk, 0, stream>>>(x, g2 + (size_t)l * Dm, be2 + (size_t)l * Dm, hbuf);
    gemm_wmma<false, false, true><<<gF, blk, 0, stream>>>(
        hbuf, W1 + (size_t)l * Dm * Fm, b1 + (size_t)l * Fm, nullptr, ffn, Mm, Fm, Dm);
    gemm_wmma<true, true, false><<<gD, blk, 0, stream>>>(
        ffn, W2 + (size_t)l * Fm * Dm, b2 + (size_t)l * Dm, x, x, Mm, Dm, Fm);
  }

  layernorm_kernel<<<Mm, blk, 0, stream>>>(x, gf, bfp, hbuf);
  gemm_wmma<true, false, false><<<gV, blk, 0, stream>>>(
      hbuf, Wout, bout, nullptr, d_out, Mm, Vm, Dm);
}
